// MultiHeadAttention_81982335746566
// MI455X (gfx1250) — matrix operand order repaired and checked by reading, not compiled, not run on hardware
//
#include <hip/hip_runtime.h>
#include <hip/hip_bf16.h>
#include <stdint.h>

// ---------------------------------------------------------------------------
// MHA for MI455X (gfx1250): bf16 WMMA path, fp32 accumulate.
//   B=2, S=4096, D_MODEL=512, H=8, Dh=64
//   GEMMs: 64x64 tile per wave (16 WMMA / K-step). Attention: 32 queries/wave.
// ---------------------------------------------------------------------------

typedef __bf16 bf16;
typedef __attribute__((ext_vector_type(8)))  __bf16 v8bf;
typedef __attribute__((ext_vector_type(16))) __bf16 v16bf;
typedef __attribute__((ext_vector_type(8)))  float  v8f;

#define MHA_B   2
#define MHA_S   4096
#define MHA_D   512
#define MHA_H   8
#define MHA_DH  64
#define MHA_M   (MHA_B * MHA_S)      // 8192 rows of activations

__device__ __forceinline__ uint16_t f2bf_bits(float f) {
    uint32_t u = __float_as_uint(f);
    u += 0x7FFFu + ((u >> 16) & 1u);          // round-to-nearest-even
    return (uint16_t)(u >> 16);
}

__device__ __forceinline__ v16bf cat16(v8bf lo, v8bf hi) {
    return __builtin_shufflevector(lo, hi, 0,1,2,3,4,5,6,7,8,9,10,11,12,13,14,15);
}

// contiguous 16 bf16 (B fragment / V^T fragment): K = 16g..16g+15, lane = N
__device__ __forceinline__ v16bf ld16(const bf16* p) {
    v8bf lo = *(const v8bf*)(p);
    v8bf hi = *(const v8bf*)(p + 8);
    return cat16(lo, hi);
}

// A fragment: contiguous 16 elements (K = 16g..16g+15)
__device__ __forceinline__ v16bf ld16_a(const bf16* row, int g) {
    v8bf lo = *(const v8bf*)(row + 16 * g);
    v8bf hi = *(const v8bf*)(row + 16 * g + 8);
    return cat16(lo, hi);
}

__device__ __forceinline__ v8f wmma_bf16(v16bf a, v16bf b, v8f c) {
    return __builtin_amdgcn_wmma_f32_16x16x32_bf16(
        /*neg_a=*/false, a, /*neg_b=*/false, b,
        /*c_mod=*/(short)0, c, /*reuse_a=*/false, /*reuse_b=*/false);
}

// ---------------------------------------------------------------------------
// Stage 0: fp32 -> bf16 conversion
// ---------------------------------------------------------------------------
__global__ void mha_cvt_f32_bf16(const float* __restrict__ src,
                                 uint16_t* __restrict__ dst, int n) {
    int i = blockIdx.x * blockDim.x + threadIdx.x;
    if (i < n) dst[i] = f2bf_bits(src[i]);
}

// ---------------------------------------------------------------------------
// Stage 1/3: GEMM  out = A[M,K](bf16) * W[N,K]^T(bf16) + bias.
//   One wave computes a 64x64 output tile: 4x4 v8f accumulators,
//   16 v_wmma per K-step of 32 against 16 b128 fragment loads.
//   mode 0: out bf16 [B,H,S,Dh]   (Q/K projection; scale folded in)
//   mode 1: out bf16 [B,H,Dh,S]   (V projection, transposed for PV fragments)
//   mode 2: out f32  [M,N]        (final output projection)
// ---------------------------------------------------------------------------
__global__ void __launch_bounds__(128)
mha_gemm_wmma(const uint16_t* __restrict__ A, const uint16_t* __restrict__ W,
              const float* __restrict__ bias, void* __restrict__ out,
              int M, int N, int K, int mode, float scale) {
    const int lane = threadIdx.x & 31;
    const int wid  = threadIdx.x >> 5;
    const int g    = lane >> 4;
    const int ln   = lane & 15;

    const int tile = blockIdx.x * (blockDim.x >> 5) + wid;
    const int ntn  = N >> 6;
    const int mt0  = (tile / ntn) << 6;
    const int nt0  = (tile % ntn) << 6;
    if (mt0 >= M) return;

    const bf16* Ab = (const bf16*)A;
    const bf16* Wb = (const bf16*)W;

    v8f acc[4][4] = {};

    for (int k0 = 0; k0 < K; k0 += 32) {
        // prefetch next K-chunk of this tile's rows (global_prefetch_b8)
        __builtin_prefetch(Ab + (size_t)(mt0 + ln) * K + k0 + 64, 0, 3);
        __builtin_prefetch(Wb + (size_t)(nt0 + ln) * K + k0 + 64, 0, 3);

        v16bf af[4], bf[4];
#pragma unroll
        for (int mt = 0; mt < 4; ++mt)
            af[mt] = ld16_a(Ab + (size_t)(mt0 + 16 * mt + ln) * K + k0, g);
#pragma unroll
        for (int nt = 0; nt < 4; ++nt)
            bf[nt] = ld16(Wb + (size_t)(nt0 + 16 * nt + ln) * K + k0 + 16 * g);
#pragma unroll
        for (int mt = 0; mt < 4; ++mt)
#pragma unroll
            for (int nt = 0; nt < 4; ++nt)
                acc[mt][nt] = wmma_bf16(af[mt], bf[nt], acc[mt][nt]);
    }

#pragma unroll
    for (int nt = 0; nt < 4; ++nt) {
        const int col = nt0 + 16 * nt + ln;
        const float bv = bias ? bias[col] : 0.0f;
#pragma unroll
        for (int mt = 0; mt < 4; ++mt) {
#pragma unroll
            for (int r = 0; r < 8; ++r) {
                const int row = mt0 + 16 * mt + r + 8 * g;
                const float v = (acc[mt][nt][r] + bv) * scale;
                if (mode == 2) {
                    ((float*)out)[(size_t)row * N + col] = v;
                } else {
                    const int batch = row >> 12, sq = row & (MHA_S - 1);
                    const int h = col >> 6, d = col & (MHA_DH - 1);
                    size_t idx;
                    if (mode == 0)
                        idx = (((size_t)batch * MHA_H + h) * MHA_S + sq) * MHA_DH + d;
                    else
                        idx = (((size_t)batch * MHA_H + h) * MHA_DH + d) * MHA_S + sq;
                    ((uint16_t*)out)[idx] = f2bf_bits(v);
                }
            }
        }
    }
}

// ---------------------------------------------------------------------------
// Stage 2: flash attention. One wave = 32 queries of one (b,h).
//   Qp/Kp: [B*H, S, 64] bf16 (Qp pre-scaled by 1/sqrt(Dh)); VpT: [B*H, 64, S].
//   Per 32-key step: 8 score WMMAs + 8 PV WMMAs; every K/V fragment is
//   reused across both query sub-tiles. P repacked via per-wave LDS (2 KB).
// ---------------------------------------------------------------------------
__global__ void __launch_bounds__(128)
mha_flash_attn(const uint16_t* __restrict__ Qp, const uint16_t* __restrict__ Kp,
               const uint16_t* __restrict__ VpT, uint16_t* __restrict__ Oout) {
    __shared__ __align__(16) uint16_t lds_p[4][2 * 16 * 32];   // 2 KB per wave

    const int lane = threadIdx.x & 31;
    const int wid  = threadIdx.x >> 5;
    const int g    = lane >> 4;
    const int ln   = lane & 15;

    const int t  = blockIdx.x * 4 + wid;      // query-tile id, 0..2047
    const int bh = t >> 7;                    // 0..15  (S/32 = 128 tiles per bh)
    const int q0 = (t & 127) << 5;            // 32-query base

    const bf16* Q  = (const bf16*)Qp  + (size_t)bh * MHA_S * MHA_DH;
    const bf16* Kb = (const bf16*)Kp  + (size_t)bh * MHA_S * MHA_DH;
    const bf16* Vt = (const bf16*)VpT + (size_t)bh * MHA_DH * MHA_S;
    uint16_t*   pl = lds_p[wid];

    // Q fragments for both 16-query sub-tiles and both Dh chunks, hoisted.
    v16bf aq[2][2];
#pragma unroll
    for (int mt = 0; mt < 2; ++mt)
#pragma unroll
        for (int kk = 0; kk < 2; ++kk)
            aq[mt][kk] = ld16_a(Q + (size_t)(q0 + 16 * mt + ln) * MHA_DH + kk * 32, g);

    v8f o[2][4] = {};
    float mrun[2][8], lrun[2][8];
#pragma unroll
    for (int mt = 0; mt < 2; ++mt)
#pragma unroll
        for (int r = 0; r < 8; ++r) { mrun[mt][r] = -3.0e38f; lrun[mt][r] = 0.0f; }

    for (int kb = 0; kb < MHA_S; kb += 32) {
        // prefetch next key tile of K and V (global_prefetch_b8)
        __builtin_prefetch(Kb + (size_t)(kb + 32 + ln) * MHA_DH, 0, 3);
        __builtin_prefetch(Vt + (size_t)(16 + ln) * MHA_S + kb + 32, 0, 3);

        // ---- scores: 32 queries x 32 keys -------------------------------
        v8f s[2][2] = {};
#pragma unroll
        for (int kk = 0; kk < 2; ++kk) {
            v16bf b0 = ld16(Kb + (size_t)(kb + ln)      * MHA_DH + kk * 32 + 16 * g);
            v16bf b1 = ld16(Kb + (size_t)(kb + 16 + ln) * MHA_DH + kk * 32 + 16 * g);
#pragma unroll
            for (int mt = 0; mt < 2; ++mt) {
                s[mt][0] = wmma_bf16(aq[mt][kk], b0, s[mt][0]);
                s[mt][1] = wmma_bf16(aq[mt][kk], b1, s[mt][1]);
            }
        }

        // ---- online softmax (row = r + 8g, cols across 16 lanes) --------
#pragma unroll
        for (int mt = 0; mt < 2; ++mt) {
            uint16_t* plm = pl + mt * (16 * 32);
#pragma unroll
            for (int r = 0; r < 8; ++r) {
                const float a0 = s[mt][0][r], a1 = s[mt][1][r];
                float mx = fmaxf(a0, a1);
                mx = fmaxf(mx, __shfl_xor(mx, 1));
                mx = fmaxf(mx, __shfl_xor(mx, 2));
                mx = fmaxf(mx, __shfl_xor(mx, 4));
                mx = fmaxf(mx, __shfl_xor(mx, 8));
                const float mnew = fmaxf(mrun[mt][r], mx);
                const float corr = __expf(mrun[mt][r] - mnew);
                const float p0 = __expf(a0 - mnew);
                const float p1 = __expf(a1 - mnew);
                float rs = p0 + p1;
                rs += __shfl_xor(rs, 1);
                rs += __shfl_xor(rs, 2);
                rs += __shfl_xor(rs, 4);
                rs += __shfl_xor(rs, 8);
                lrun[mt][r] = lrun[mt][r] * corr + rs;
                mrun[mt][r] = mnew;
                o[mt][0][r] *= corr; o[mt][1][r] *= corr;
                o[mt][2][r] *= corr; o[mt][3][r] *= corr;

                const int row = r + 8 * g;
                plm[row * 32 + ln]      = f2bf_bits(p0);
                plm[row * 32 + 16 + ln] = f2bf_bits(p1);
            }
        }
        __syncthreads();

        // ---- P: C-layout -> A-fragment (lane = query row, elems = keys) --
        v16bf pa[2];
#pragma unroll
        for (int mt = 0; mt < 2; ++mt)
            pa[mt] = ld16_a((const bf16*)(pl + mt * (16 * 32) + ln * 32), g);

        // ---- O += P * V  (N-tiles over Dh; V fragment shared by both mt) -
#pragma unroll
        for (int nt = 0; nt < 4; ++nt) {
            v16bf bv = ld16(Vt + (size_t)(nt * 16 + ln) * MHA_S + kb + 16 * g);
#pragma unroll
            for (int mt = 0; mt < 2; ++mt)
                o[mt][nt] = wmma_bf16(pa[mt], bv, o[mt][nt]);
        }
        __syncthreads();
    }

    // ---- normalize + write back in model layout [B*S, 512] bf16 ---------
    const int batch = bh >> 3, h = bh & 7;
#pragma unroll
    for (int mt = 0; mt < 2; ++mt) {
#pragma unroll
        for (int r = 0; r < 8; ++r) {
            const float inv = 1.0f / lrun[mt][r];
            const int row = q0 + 16 * mt + r + 8 * g;
#pragma unroll
            for (int nt = 0; nt < 4; ++nt) {
                const size_t idx = ((size_t)batch * MHA_S + row) * MHA_D
                                 + h * MHA_DH + nt * 16 + ln;
                Oout[idx] = f2bf_bits(o[mt][nt][r] * inv);
            }
        }
    }
}

// ---------------------------------------------------------------------------
// Host launcher
// ---------------------------------------------------------------------------
extern "C" void kernel_launch(void* const* d_in, const int* in_sizes, int n_in,
                              void* d_out, int out_size, void* d_ws, size_t ws_size,
                              hipStream_t stream) {
    const float* q  = (const float*)d_in[0];
    const float* k  = (const float*)d_in[1];
    const float* v  = (const float*)d_in[2];
    const float* Wq = (const float*)d_in[3];
    const float* bq = (const float*)d_in[4];
    const float* Wk = (const float*)d_in[5];
    const float* bk = (const float*)d_in[6];
    const float* Wv = (const float*)d_in[7];
    const float* bv = (const float*)d_in[8];
    const float* Wo = (const float*)d_in[9];
    const float* bo = (const float*)d_in[10];

    const size_t ACT = (size_t)MHA_M * MHA_D * 2;  // 8 MiB bf16 activation buf
    const size_t WGT = (size_t)MHA_D * MHA_D * 2;  // 512 KiB bf16 weight buf
    uint8_t* ws = (uint8_t*)d_ws;
    uint16_t* qb   = (uint16_t*)(ws);
    uint16_t* kb   = (uint16_t*)(ws + 1 * ACT);
    uint16_t* vb   = (uint16_t*)(ws + 2 * ACT);
    uint16_t* Wqb  = (uint16_t*)(ws + 3 * ACT);
    uint16_t* Wkb  = (uint16_t*)(ws + 3 * ACT + 1 * WGT);
    uint16_t* Wvb  = (uint16_t*)(ws + 3 * ACT + 2 * WGT);
    uint16_t* Wob  = (uint16_t*)(ws + 3 * ACT + 3 * WGT);
    uint16_t* Qp   = (uint16_t*)(ws + 3 * ACT + 4 * WGT);
    uint16_t* Kp   = (uint16_t*)(ws + 4 * ACT + 4 * WGT);
    uint16_t* VpT  = (uint16_t*)(ws + 5 * ACT + 4 * WGT);
    uint16_t* AOut = (uint16_t*)(ws + 6 * ACT + 4 * WGT);

    const int n_act = MHA_M * MHA_D;        // 4194304
    const int n_wgt = MHA_D * MHA_D;        // 262144
    mha_cvt_f32_bf16<<<(n_act + 255) / 256, 256, 0, stream>>>(q,  qb,  n_act);
    mha_cvt_f32_bf16<<<(n_act + 255) / 256, 256, 0, stream>>>(k,  kb,  n_act);
    mha_cvt_f32_bf16<<<(n_act + 255) / 256, 256, 0, stream>>>(v,  vb,  n_act);
    mha_cvt_f32_bf16<<<(n_wgt + 255) / 256, 256, 0, stream>>>(Wq, Wqb, n_wgt);
    mha_cvt_f32_bf16<<<(n_wgt + 255) / 256, 256, 0, stream>>>(Wk, Wkb, n_wgt);
    mha_cvt_f32_bf16<<<(n_wgt + 255) / 256, 256, 0, stream>>>(Wv, Wvb, n_wgt);
    mha_cvt_f32_bf16<<<(n_wgt + 255) / 256, 256, 0, stream>>>(Wo, Wob, n_wgt);

    const int tiles = (MHA_M / 64) * (MHA_D / 64);   // 1024 wave-tiles (64x64)
    const dim3 gblk(128);
    const float inv_sqrt_dh = 0.125f;                // 1/sqrt(64)
    mha_gemm_wmma<<<tiles / 4, gblk, 0, stream>>>(qb, Wqb, bq, Qp,
        MHA_M, MHA_D, MHA_D, /*mode=*/0, inv_sqrt_dh);
    mha_gemm_wmma<<<tiles / 4, gblk, 0, stream>>>(kb, Wkb, bk, Kp,
        MHA_M, MHA_D, MHA_D, /*mode=*/0, 1.0f);
    mha_gemm_wmma<<<tiles / 4, gblk, 0, stream>>>(vb, Wvb, bv, VpT,
        MHA_M, MHA_D, MHA_D, /*mode=*/1, 1.0f);

    const int qtiles = MHA_B * MHA_H * (MHA_S / 32); // 2048 (32-query tiles)
    mha_flash_attn<<<qtiles / 4, 128, 0, stream>>>(Qp, Kp, VpT, AOut);

    mha_gemm_wmma<<<tiles / 4, gblk, 0, stream>>>(AOut, Wob, bo, (float*)d_out,
        MHA_M, MHA_D, MHA_D, /*mode=*/2, 1.0f);
}
